// SharedQKVAttention_71614284694165
// MI455X (gfx1250) — matrix operand order repaired and checked by reading, not compiled, not run on hardware
//
#include <hip/hip_runtime.h>
#include <hip/hip_bf16.h>
#include <math.h>

typedef __bf16 bf16;
typedef __attribute__((ext_vector_type(16))) __bf16 v16bf;
typedef __attribute__((ext_vector_type(8)))  __bf16 v8bf;
typedef __attribute__((ext_vector_type(4)))  __bf16 v4bf;
typedef __attribute__((ext_vector_type(8)))  float  v8f;
typedef int v4i __attribute__((__vector_size__(16)));

static constexpr int Bb = 4, Nn = 2048, Ff = 1024, Hh = 16, Dh = 64;
// exp2 argument scale: (1/sqrt(64)) * log2(e)
static constexpr float CEXP = 0.125f * 1.4426950408889634f;

union Frag {
    v16bf v;
    v8bf  p[2];
};

__device__ __forceinline__ v8f wmma_bf16(const Frag& a, const Frag& b, v8f c) {
    return __builtin_amdgcn_wmma_f32_16x16x32_bf16(
        false, a.v, false, b.v, (short)0, c, false, false);
}

// ---------------------------------------------------------------------------
// CDNA5 async global->LDS staging (ASYNCcnt path), with plain-load fallback.
// Signature verified by compile probe: (AS1 v4i*, AS3 v4i*, imm off, imm cpol)
// ---------------------------------------------------------------------------
#if __has_builtin(__builtin_amdgcn_global_load_async_to_lds_b128)
#define HAVE_ASYNC_LDS 1
#else
#define HAVE_ASYNC_LDS 0
#endif

__device__ __forceinline__ void stage16(const bf16* g, bf16* l) {
#if HAVE_ASYNC_LDS
    __builtin_amdgcn_global_load_async_to_lds_b128(
        (__attribute__((address_space(1))) v4i*)(unsigned long long)(g),
        (__attribute__((address_space(3))) v4i*)(unsigned int)(unsigned long long)(l),
        0, 0);
#else
    *(v8bf*)l = *(const v8bf*)g;
#endif
}

__device__ __forceinline__ void wait_async_lds() {
#if HAVE_ASYNC_LDS
#if __has_builtin(__builtin_amdgcn_s_wait_asynccnt)
    __builtin_amdgcn_s_wait_asynccnt(0);
#else
    asm volatile("s_wait_asynccnt 0x0" ::: "memory");
#endif
#endif
}

// ---------------------------------------------------------------------------
// 16-lane butterfly max: v_permlane16_b32 (VALU pipe) if available,
// else ds_bpermute via __shfl_xor (LDS pipe).
// Nibble tables encode sel[i] = i ^ mask for lanes 0-7 (lo) / 8-15 (hi).
// ---------------------------------------------------------------------------
#if __has_builtin(__builtin_amdgcn_permlane16)
#define HAVE_PERMLANE16 1
#else
#define HAVE_PERMLANE16 0
#endif

__device__ __forceinline__ float xmax16(float t) {
#if HAVE_PERMLANE16
    int x = __float_as_int(t);
    int y;
    y = __builtin_amdgcn_permlane16(x, x, 0x67452301u, 0xEFCDAB89u, false, false); // ^1
    x = __float_as_int(fmaxf(__int_as_float(x), __int_as_float(y)));
    y = __builtin_amdgcn_permlane16(x, x, 0x54761032u, 0xDCFE98BAu, false, false); // ^2
    x = __float_as_int(fmaxf(__int_as_float(x), __int_as_float(y)));
    y = __builtin_amdgcn_permlane16(x, x, 0x32107654u, 0xBA98FEDCu, false, false); // ^4
    x = __float_as_int(fmaxf(__int_as_float(x), __int_as_float(y)));
    y = __builtin_amdgcn_permlane16(x, x, 0xFEDCBA98u, 0x76543210u, false, false); // ^8
    return fmaxf(__int_as_float(x), __int_as_float(y));
#else
    t = fmaxf(t, __shfl_xor(t, 1));
    t = fmaxf(t, __shfl_xor(t, 2));
    t = fmaxf(t, __shfl_xor(t, 4));
    t = fmaxf(t, __shfl_xor(t, 8));
    return t;
#endif
}

// ---------------------------------------------------------------------------
// Kernel 0: fp32 -> bf16 conversion
// ---------------------------------------------------------------------------
__global__ __launch_bounds__(256) void cvt_f32_to_bf16(
    const float* __restrict__ src, bf16* __restrict__ dst, int n)
{
    int i = (blockIdx.x * blockDim.x + threadIdx.x) * 4;
    if (i >= n) return;
    float4 f = *(const float4*)(src + i);
    v4bf r;
    r[0] = (bf16)f.x; r[1] = (bf16)f.y; r[2] = (bf16)f.z; r[3] = (bf16)f.w;
    *(v4bf*)(dst + i) = r;
}

// ---------------------------------------------------------------------------
// GEMM (NT): C[m][o] = sum_k A[m][k] * Bm[o][k]
// MODE 0: bf16 store into qkv [B][H][N][DH];  MODE 1: f32 row-major store
// 256 thr = 8 waves, wave tile 64x64 (4x4 WMMA), block tile 128x256
// ---------------------------------------------------------------------------
template<int MODE>
__global__ __launch_bounds__(256) void gemm_bf16_nt(
    const bf16* __restrict__ A, const bf16* __restrict__ Bm,
    bf16* __restrict__ outQ, float* __restrict__ outF,
    int Ncols, int K)
{
    const int lane = threadIdx.x & 31;
    const int wave = threadIdx.x >> 5;
    const int hl   = lane >> 4;
    const int l16  = lane & 15;
    const int wm   = wave >> 2;
    const int wn   = wave & 3;
    const int m_base = blockIdx.x * 128 + wm * 64;
    const int n_base = blockIdx.y * 256 + wn * 64;

    const bf16* aRow[4];
    const bf16* bRow[4];
#pragma unroll
    for (int i = 0; i < 4; ++i) aRow[i] = A  + (size_t)(m_base + i * 16 + l16) * K;
#pragma unroll
    for (int j = 0; j < 4; ++j) bRow[j] = Bm + (size_t)(n_base + j * 16 + l16) * K;

    v8f c[4][4];
#pragma unroll
    for (int i = 0; i < 4; ++i)
#pragma unroll
        for (int j = 0; j < 4; ++j) c[i][j] = (v8f){0.f,0.f,0.f,0.f,0.f,0.f,0.f,0.f};

    for (int k0 = 0; k0 < K; k0 += 32) {
        Frag af[4], bfr[4];
#pragma unroll
        for (int i = 0; i < 4; ++i) {
            af[i].p[0] = *(const v8bf*)(aRow[i] + k0 + 16 * hl);
            af[i].p[1] = *(const v8bf*)(aRow[i] + k0 + 16 * hl + 8);
        }
#pragma unroll
        for (int j = 0; j < 4; ++j) {
            bfr[j].p[0] = *(const v8bf*)(bRow[j] + k0 + 16 * hl);
            bfr[j].p[1] = *(const v8bf*)(bRow[j] + k0 + 16 * hl + 8);
        }
#pragma unroll
        for (int i = 0; i < 4; ++i)
#pragma unroll
            for (int j = 0; j < 4; ++j)
                c[i][j] = wmma_bf16(af[i], bfr[j], c[i][j]);
    }

#pragma unroll
    for (int i = 0; i < 4; ++i) {
#pragma unroll
        for (int j = 0; j < 4; ++j) {
#pragma unroll
            for (int r = 0; r < 8; ++r) {
                int row = m_base + i * 16 + r + 8 * hl;
                int col = n_base + j * 16 + l16;
                float v = c[i][j][r];
                if (MODE == 0) {
                    int bi = row >> 11;
                    int nn = row & (Nn - 1);
                    int hh = col >> 6;
                    int dh = col & (Dh - 1);
                    outQ[((size_t)(bi * Hh + hh) * Nn + nn) * Dh + dh] = (bf16)v;
                } else {
                    outF[(size_t)row * Ncols + col] = v;
                }
            }
        }
    }
}

// ---------------------------------------------------------------------------
// Attention, flash-style, key blocks of 64.
// Block = 8 waves x 16 queries = 128.  Q=K=V (shared qkv).
// Row sums via P @ ones WMMA; running max via permlane16 butterfly.
// ---------------------------------------------------------------------------
__global__ __launch_bounds__(256) void attn_flash_bf16(
    const bf16* __restrict__ Q, bf16* __restrict__ Aout)
{
    const int lane = threadIdx.x & 31;
    const int wave = threadIdx.x >> 5;
    const int hl   = lane >> 4;
    const int l16  = lane & 15;
    const int bh   = blockIdx.y;
    const int bi   = bh >> 4;
    const int hh   = bh & 15;
    const int m0   = blockIdx.x * 128 + wave * 16;

    const bf16* Qbh = Q + (size_t)bh * Nn * Dh;     // [Nn][Dh]

    __shared__ __align__(16) bf16 kvN[64 * 64];         // [key][dh]
    __shared__ __align__(16) bf16 kvT[64 * 80];         // [dh][key], stride 80
    __shared__ __align__(16) bf16 pbuf[8][16 * 64];     // per-wave P tile

    // Query A-fragments (k = dh, 2 steps of 32), resident in VGPRs
    Frag qf[2];
    {
        const bf16* qrow = Qbh + (size_t)(m0 + l16) * Dh;
#pragma unroll
        for (int kk = 0; kk < 2; ++kk) {
            qf[kk].p[0] = *(const v8bf*)(qrow + kk * 32 + 16 * hl);
            qf[kk].p[1] = *(const v8bf*)(qrow + kk * 32 + 16 * hl + 8);
        }
    }

    // all-ones B fragment for row-sum WMMA
    Frag onesf;
#pragma unroll
    for (int e = 0; e < 16; ++e) onesf.v[e] = (bf16)1.0f;

    v8f o[4];
#pragma unroll
    for (int jt = 0; jt < 4; ++jt) o[jt] = (v8f){0.f,0.f,0.f,0.f,0.f,0.f,0.f,0.f};
    float mrow[8], lrow[8];
#pragma unroll
    for (int r = 0; r < 8; ++r) { mrow[r] = -INFINITY; lrow[r] = 0.f; }

    const int tKey = threadIdx.x >> 2;        // 0..63
    const int tCol = (threadIdx.x & 3) * 16;  // 0,16,32,48

    for (int kb = 0; kb < Nn; kb += 64) {
        // ---- stage KV tile [64 keys x 64 dh] into LDS (async path) ----
        const bf16* gsrc = Qbh + (size_t)(kb + tKey) * Dh + tCol;
        bf16* ld = kvN + tKey * 64 + tCol;
        stage16(gsrc,     ld);
        stage16(gsrc + 8, ld + 8);
        wait_async_lds();
        // build transposed copy from our own freshly-staged chunk
        v8bf c0 = *(const v8bf*)ld;
        v8bf c1 = *(const v8bf*)(ld + 8);
#pragma unroll
        for (int e = 0; e < 8; ++e) {
            kvT[(tCol + e)     * 80 + tKey] = c0[e];
            kvT[(tCol + 8 + e) * 80 + tKey] = c1[e];
        }
        __syncthreads();

        // ---- S = Q * K^T : four 16-key tiles, unscaled scores ----
        v8f s[4];
#pragma unroll
        for (int j = 0; j < 4; ++j) s[j] = (v8f){0.f,0.f,0.f,0.f,0.f,0.f,0.f,0.f};
#pragma unroll
        for (int j = 0; j < 4; ++j) {
#pragma unroll
            for (int kk = 0; kk < 2; ++kk) {
                Frag bK;
                const bf16* p = kvN + (j * 16 + l16) * 64 + kk * 32 + 16 * hl;
                bK.p[0] = *(const v8bf*)(p);
                bK.p[1] = *(const v8bf*)(p + 8);
                s[j] = wmma_bf16(qf[kk], bK, s[j]);
            }
        }

        // ---- running max (scale folded into exp argument) ----
        float newm[8], alpha[8];
#pragma unroll
        for (int r = 0; r < 8; ++r) {
            float t = fmaxf(fmaxf(s[0][r], s[1][r]), fmaxf(s[2][r], s[3][r]));
            t = xmax16(t);
            newm[r]  = fmaxf(mrow[r], t);
            alpha[r] = exp2f((mrow[r] - newm[r]) * CEXP);
            mrow[r]  = newm[r];
        }

        // ---- P = exp2((S - m) * c), write to per-wave LDS (relayout) ----
        bf16* pw = &pbuf[wave][0];
#pragma unroll
        for (int j = 0; j < 4; ++j)
#pragma unroll
            for (int r = 0; r < 8; ++r)
                pw[(r + 8 * hl) * 64 + j * 16 + l16] =
                    (bf16)exp2f((s[j][r] - newm[r]) * CEXP);

        // ---- reload P as A-fragments (two k-steps of 32 keys) ----
        Frag pf[2];
#pragma unroll
        for (int kk = 0; kk < 2; ++kk) {
            pf[kk].p[0] = *(const v8bf*)(pw + l16 * 64 + kk * 32 + 16 * hl);
            pf[kk].p[1] = *(const v8bf*)(pw + l16 * 64 + kk * 32 + 16 * hl + 8);
        }

        // ---- row sums via P @ ones (replicated to all lanes) ----
        v8f rs = (v8f){0.f,0.f,0.f,0.f,0.f,0.f,0.f,0.f};
        rs = wmma_bf16(pf[0], onesf, rs);
        rs = wmma_bf16(pf[1], onesf, rs);
#pragma unroll
        for (int r = 0; r < 8; ++r) lrow[r] = lrow[r] * alpha[r] + rs[r];

        // ---- rescale O, then O += P @ V ----
#pragma unroll
        for (int jt = 0; jt < 4; ++jt)
#pragma unroll
            for (int r = 0; r < 8; ++r) o[jt][r] *= alpha[r];

#pragma unroll
        for (int kk = 0; kk < 2; ++kk) {
#pragma unroll
            for (int jt = 0; jt < 4; ++jt) {
                Frag bV;
                const bf16* p = kvT + (jt * 16 + l16) * 80 + kk * 32 + 16 * hl;
                bV.p[0] = *(const v8bf*)(p);
                bV.p[1] = *(const v8bf*)(p + 8);
                o[jt] = wmma_bf16(pf[kk], bV, o[jt]);
            }
        }
        __syncthreads();
    }

    // ---- epilogue: O /= l, store bf16 into attn [B*N][F] ----
#pragma unroll
    for (int r = 0; r < 8; ++r) {
        float inv = 1.0f / lrow[r];
        int row = m0 + r + 8 * hl;
#pragma unroll
        for (int jt = 0; jt < 4; ++jt) {
            float v = o[jt][r] * inv;
            int col = hh * Dh + jt * 16 + l16;
            Aout[(size_t)(bi * Nn + row) * Ff + col] = (bf16)v;
        }
    }
}

// ---------------------------------------------------------------------------
// Host-side launch
// ---------------------------------------------------------------------------
extern "C" void kernel_launch(void* const* d_in, const int* in_sizes, int n_in,
                              void* d_out, int out_size, void* d_ws, size_t ws_size,
                              hipStream_t stream) {
    const float* x    = (const float*)d_in[0];
    const float* wqkv = (const float*)d_in[1];
    const float* wo   = (const float*)d_in[2];
    float*       out  = (float*)d_out;

    char* ws = (char*)d_ws;
    const size_t szX = (size_t)Bb * Nn * Ff * 2;
    const size_t szW = (size_t)Ff * Ff * 2;
    bf16* Xb  = (bf16*)(ws);
    bf16* Wqb = (bf16*)(ws + szX);
    bf16* Wob = (bf16*)(ws + szX + szW);
    bf16* Qkv = (bf16*)(ws + szX + 2 * szW);
    bf16* Att = (bf16*)(ws + 2 * szX + 2 * szW);

    const int nx = Bb * Nn * Ff;
    const int nw = Ff * Ff;
    cvt_f32_to_bf16<<<nx / (4 * 256), 256, 0, stream>>>(x,    Xb,  nx);
    cvt_f32_to_bf16<<<nw / (4 * 256), 256, 0, stream>>>(wqkv, Wqb, nw);
    cvt_f32_to_bf16<<<nw / (4 * 256), 256, 0, stream>>>(wo,   Wob, nw);

    dim3 gGemm(8192 / 128, 1024 / 256);
    gemm_bf16_nt<0><<<gGemm, 256, 0, stream>>>(Xb, Wqb, Qkv, nullptr, 1024, 1024);

    dim3 gAttn(Nn / 128, Bb * Hh);
    attn_flash_bf16<<<gAttn, 256, 0, stream>>>(Qkv, Att);

    gemm_bf16_nt<1><<<gGemm, 256, 0, stream>>>(Att, Wob, nullptr, out, 1024, 1024);
}
